// SparseMoE_37623913513502
// MI455X (gfx1250) — matrix operand order repaired and checked by reading, not compiled, not run on hardware
//
#include <hip/hip_runtime.h>
#include <hip/hip_bf16.h>

#define BQ 4
#define SQ 1024
#define DQ 512
#define EQ 10
#define HQ 2048
#define NTOK (BQ*SQ)   // 4096 tokens
#define MT 32          // tokens per FFN block (2 WMMA M-subtiles)

typedef __bf16 bf16;
typedef __attribute__((ext_vector_type(8)))  __bf16 v8bf;
typedef __attribute__((ext_vector_type(16))) __bf16 v16bf;
typedef __attribute__((ext_vector_type(8)))  float  v8f;

// tanh-approx gelu in sigmoid form: gelu(x) = x * sigmoid(2*c0*(x + c1*x^3)).
// Branch-free: v_exp_f32 + v_rcp_f32.  Stable at both tails.
__device__ __forceinline__ float gelu_fast(float x){
    const float k  = 1.5957691216057308f;            // 2*sqrt(2/pi)
    const float kc = 1.5957691216057308f * 0.044715f;
    float x2  = x * x;
    float arg = -(k * x + kc * x * x2);              // -2*z
    float den = 1.0f + __expf(arg);
    return x * __builtin_amdgcn_rcpf(den);
}

// ---------------- zero output + router accumulators ----------------
__global__ void moe_zero_kernel(float* __restrict__ out, float* __restrict__ acc, int n){
    int i = blockIdx.x * blockDim.x + threadIdx.x;
    if (i < n) out[i] = 0.0f;
    if (i < 2*EQ) acc[i] = 0.0f;
}

// ---------------- router: softmax + top-2 gates + loss sums ----------------
__global__ void moe_router_kernel(const float* __restrict__ x, const float* __restrict__ Wr,
                                  const float* __restrict__ br, float* __restrict__ gates,
                                  float* __restrict__ acc /* [0..9]=imp sum, [10..19]=load sum */){
    __shared__ float red[64][EQ];
    const int tok = blockIdx.x;
    const int tid = threadIdx.x;
    float a[EQ];
#pragma unroll
    for (int e = 0; e < EQ; ++e) a[e] = 0.0f;
    const float* xr = x + (size_t)tok * DQ;
    for (int d = tid; d < DQ; d += 64){
        float xv = xr[d];
        const float* w = Wr + (size_t)d * EQ;   // Wr is [D][E], e contiguous
#pragma unroll
        for (int e = 0; e < EQ; ++e) a[e] += xv * w[e];
    }
#pragma unroll
    for (int e = 0; e < EQ; ++e) red[tid][e] = a[e];
    __syncthreads();
    if (tid == 0){
        float logit[EQ];
        for (int e = 0; e < EQ; ++e){
            float s = br[e];
            for (int t = 0; t < 64; ++t) s += red[t][e];
            logit[e] = s;
        }
        float m = logit[0];
        for (int e = 1; e < EQ; ++e) m = fmaxf(m, logit[e]);
        float p[EQ]; float sum = 0.0f;
        for (int e = 0; e < EQ; ++e){ p[e] = __expf(logit[e] - m); sum += p[e]; }
        float inv = 1.0f / sum;
        for (int e = 0; e < EQ; ++e) p[e] *= inv;
        // top-2 (ties -> lowest index, matches lax.top_k)
        int i1 = 0;
        for (int e = 1; e < EQ; ++e) if (p[e] > p[i1]) i1 = e;
        int i2 = (i1 == 0) ? 1 : 0;
        for (int e = 0; e < EQ; ++e) if (e != i1 && p[e] > p[i2]) i2 = e;
        for (int e = 0; e < EQ; ++e){
            gates[(size_t)tok*EQ + e] = (e == i1 || e == i2) ? p[e] : 0.0f;
            atomicAdd(&acc[e], p[e]);             // importance sum
        }
        atomicAdd(&acc[EQ + i1], 1.0f);           // load counts
        atomicAdd(&acc[EQ + i2], 1.0f);
    }
}

__global__ void moe_loss_kernel(const float* __restrict__ acc, float* __restrict__ out_loss){
    float s = 0.0f;
    const float invN = 1.0f / (float)NTOK;
    for (int e = 0; e < EQ; ++e) s += (acc[e]*invN) * (acc[EQ+e]*invN);
    *out_loss = (float)EQ * s;
}

// ---------------- fp32 -> bf16 convert / transpose ----------------
__global__ void moe_cvt_kernel(const float* __restrict__ in, bf16* __restrict__ out, int n){
    int i = blockIdx.x * blockDim.x + threadIdx.x;
    if (i < n) out[i] = (bf16)in[i];
}

// in: [E][rows][cols] fp32 row-major  ->  out: [E][cols][rows] bf16
__global__ void moe_transpose_kernel(const float* __restrict__ in, bf16* __restrict__ out,
                                     int rows, int cols){
    size_t i = (size_t)blockIdx.x * blockDim.x + threadIdx.x;
    size_t total = (size_t)EQ * rows * cols;
    if (i >= total) return;
    int c = (int)(i % cols);
    size_t t = i / cols;
    int r = (int)(t % rows);
    int e = (int)(t / rows);
    out[((size_t)e * cols + c) * rows + r] = (bf16)in[i];
}

// ---------------- fused dense-MoE FFN tile kernel ----------------
// grid = (NTOK/MT, E), block = 256 (8 waves).  Per block: 32-token tile x one expert.
// Two 16-row M-subtiles share every B fragment.  K-loops capped at unroll 4 to keep
// VGPR pressure low -> high wave occupancy does the memory-latency hiding.
__global__ __launch_bounds__(256) void moe_ffn_kernel(
    const bf16*  __restrict__ xb,    // [N][D] bf16
    const bf16*  __restrict__ w1t,   // [E][H][D] bf16  (W1 transposed)
    const bf16*  __restrict__ w2t,   // [E][D][H] bf16  (W2 transposed)
    const float* __restrict__ b1,    // [E][H]
    const float* __restrict__ b2,    // [E][D]
    const float* __restrict__ gates, // [N][E]
    float*       __restrict__ out)   // [N][D] (atomic accumulate)
{
    __shared__ bf16 hbuf[MT * HQ];   // 128 KB gate-scaled gelu(h) tile, bf16

    const int e     = blockIdx.y;
    const int tok0  = blockIdx.x * MT;
    const int tid   = threadIdx.x;
    const int wave  = tid >> 5;
    const int lane  = tid & 31;
    const int lhalf = lane >> 4;     // 0: lanes 0-15, 1: lanes 16-31
    const int lrow  = lane & 15;

    // skip tile if no token routed to this expert (uniform across block)
    float gsum = 0.0f;
    for (int t = 0; t < MT; ++t) gsum += fabsf(gates[(size_t)(tok0 + t)*EQ + e]);
    if (gsum == 0.0f) return;

    // gates for the C-matrix rows this lane owns in each M-subtile
    float gv0[8], gv1[8];
#pragma unroll
    for (int r = 0; r < 8; ++r){
        gv0[r] = gates[(size_t)(tok0 +      r + 8*lhalf)*EQ + e];
        gv1[r] = gates[(size_t)(tok0 + 16 + r + 8*lhalf)*EQ + e];
    }

    const bf16*  w1e = w1t + (size_t)e * HQ * DQ;
    const bf16*  w2e = w2t + (size_t)e * DQ * HQ;
    const float* b1e = b1  + (size_t)e * HQ;
    const float* b2e = b2  + (size_t)e * DQ;

    // ---- Stage 1: h = gate * gelu(x @ W1 + b1) -> bf16 LDS tile ----
    const bf16* arow0 = xb + (size_t)(tok0 +      lrow) * DQ;
    const bf16* arow1 = xb + (size_t)(tok0 + 16 + lrow) * DQ;
    for (int nt = wave; nt < HQ/16; nt += 8){
        const int nb = nt * 16;
        const float bias = b1e[nb + lrow];   // issued before the GEMM, in flight behind it
        v8f c0 = {}, c1 = {};
        const bf16* brow = w1e + (size_t)(nb + lrow) * DQ;   // B column (contig. in K)
#pragma unroll 4
        for (int kb = 0; kb < DQ; kb += 32){
            v16bf a0, a1, b;
            const v8bf al0 = *(const v8bf*)(arow0 + kb + lhalf*8);
            const v8bf ah0 = *(const v8bf*)(arow0 + kb + 16 + lhalf*8);
            const v8bf al1 = *(const v8bf*)(arow1 + kb + lhalf*8);
            const v8bf ah1 = *(const v8bf*)(arow1 + kb + 16 + lhalf*8);
            const v8bf bl  = *(const v8bf*)(brow + kb + lhalf*8);
            const v8bf bh  = *(const v8bf*)(brow + kb + 16 + lhalf*8);
#pragma unroll
            for (int i = 0; i < 8; ++i){
                a0[i]=al0[i]; a0[i+8]=ah0[i];
                a1[i]=al1[i]; a1[i+8]=ah1[i];
                b[i]=bl[i];   b[i+8]=bh[i];
            }
            c0 = __builtin_amdgcn_wmma_f32_16x16x32_bf16(false, a0, false, b,
                                                         (short)0, c0, false, false);
            c1 = __builtin_amdgcn_wmma_f32_16x16x32_bf16(false, a1, false, b,
                                                         (short)0, c1, false, false);
        }
#pragma unroll
        for (int r = 0; r < 8; ++r){
            const int row0 = r + 8*lhalf;
            hbuf[(size_t)row0        * HQ + nb + lrow] = (bf16)(gelu_fast(c0[r] + bias) * gv0[r]);
            hbuf[(size_t)(row0 + 16) * HQ + nb + lrow] = (bf16)(gelu_fast(c1[r] + bias) * gv1[r]);
        }
    }
    __syncthreads();

    // ---- Stage 2: out += h @ W2 + gate*b2 ----
    const bf16* hrow0 = hbuf + (size_t)lrow * HQ;
    const bf16* hrow1 = hbuf + (size_t)(16 + lrow) * HQ;
    for (int nt = wave; nt < DQ/16; nt += 8){
        const int nb = nt * 16;
        const float bias = b2e[nb + lrow];   // in flight behind the GEMM
        v8f c0 = {}, c1 = {};
        const bf16* brow = w2e + (size_t)(nb + lrow) * HQ;
#pragma unroll 4
        for (int kb = 0; kb < HQ; kb += 32){
            v16bf a0, a1, b;
            const v8bf al0 = *(const v8bf*)(hrow0 + kb + lhalf*8);
            const v8bf ah0 = *(const v8bf*)(hrow0 + kb + 16 + lhalf*8);
            const v8bf al1 = *(const v8bf*)(hrow1 + kb + lhalf*8);
            const v8bf ah1 = *(const v8bf*)(hrow1 + kb + 16 + lhalf*8);
            const v8bf bl  = *(const v8bf*)(brow + kb + lhalf*8);
            const v8bf bh  = *(const v8bf*)(brow + kb + 16 + lhalf*8);
#pragma unroll
            for (int i = 0; i < 8; ++i){
                a0[i]=al0[i]; a0[i+8]=ah0[i];
                a1[i]=al1[i]; a1[i+8]=ah1[i];
                b[i]=bl[i];   b[i+8]=bh[i];
            }
            c0 = __builtin_amdgcn_wmma_f32_16x16x32_bf16(false, a0, false, b,
                                                         (short)0, c0, false, false);
            c1 = __builtin_amdgcn_wmma_f32_16x16x32_bf16(false, a1, false, b,
                                                         (short)0, c1, false, false);
        }
#pragma unroll
        for (int r = 0; r < 8; ++r){
            const int row0 = r + 8*lhalf;
            float g0 = gv0[r];
            if (g0 != 0.0f){
                int token = tok0 + row0;
                atomicAdd(&out[(size_t)token * DQ + nb + lrow], c0[r] + g0 * bias);
            }
            float g1 = gv1[r];
            if (g1 != 0.0f){
                int token = tok0 + 16 + row0;
                atomicAdd(&out[(size_t)token * DQ + nb + lrow], c1[r] + g1 * bias);
            }
        }
    }
}

// ---------------- host launcher ----------------
extern "C" void kernel_launch(void* const* d_in, const int* in_sizes, int n_in,
                              void* d_out, int out_size, void* d_ws, size_t ws_size,
                              hipStream_t stream) {
    const float* x  = (const float*)d_in[0];
    const float* Wr = (const float*)d_in[1];
    const float* br = (const float*)d_in[2];
    const float* W1 = (const float*)d_in[3];
    const float* b1 = (const float*)d_in[4];
    const float* W2 = (const float*)d_in[5];
    const float* b2 = (const float*)d_in[6];
    float* out = (float*)d_out;
    (void)in_sizes; (void)n_in; (void)out_size; (void)ws_size;

    char* base = (char*)d_ws;
    size_t off = 0;
    auto take = [&](size_t bytes) -> char* {
        char* p = base + off;
        off = (off + bytes + 255) & ~(size_t)255;
        return p;
    };
    float* gates = (float*)take((size_t)NTOK * EQ * sizeof(float));
    float* acc   = (float*)take(2 * EQ * sizeof(float));
    bf16*  xb    = (bf16*) take((size_t)NTOK * DQ * sizeof(bf16));
    bf16*  w1t   = (bf16*) take((size_t)EQ * DQ * HQ * sizeof(bf16));
    bf16*  w2t   = (bf16*) take((size_t)EQ * DQ * HQ * sizeof(bf16));

    const int nOut = NTOK * DQ + 1;
    moe_zero_kernel<<<(nOut + 255)/256, 256, 0, stream>>>(out, acc, nOut);
    moe_router_kernel<<<NTOK, 64, 0, stream>>>(x, Wr, br, gates, acc);
    moe_loss_kernel<<<1, 1, 0, stream>>>(acc, out + (size_t)NTOK * DQ);

    moe_cvt_kernel<<<(NTOK*DQ + 255)/256, 256, 0, stream>>>(x, xb, NTOK*DQ);
    const int wtotal = EQ * DQ * HQ;
    moe_transpose_kernel<<<(wtotal + 255)/256, 256, 0, stream>>>(W1, w1t, DQ, HQ); // [E][D][H]->[E][H][D]
    moe_transpose_kernel<<<(wtotal + 255)/256, 256, 0, stream>>>(W2, w2t, HQ, DQ); // [E][H][D]->[E][D][H]

    dim3 grid(NTOK/MT, EQ);
    moe_ffn_kernel<<<grid, 256, 0, stream>>>(xb, w1t, w2t, b1, b2, gates, out);
}
